// Mixtral_12859132084357
// MI455X (gfx1250) — matrix operand order repaired and checked by reading, not compiled, not run on hardware
//
#include <hip/hip_runtime.h>
#include <hip/hip_bf16.h>
#include <math.h>

typedef __attribute__((ext_vector_type(16))) __bf16 bf16x16;
typedef __attribute__((ext_vector_type(8)))  __bf16 bf16x8;
typedef __attribute__((ext_vector_type(4)))  __bf16 bf16x4;
typedef __attribute__((ext_vector_type(8)))  float  f32x8;

#define TC   2048   // B*S tokens
#define BC   2
#define SC   1024
#define DC   1024
#define HC   16
#define KVHC 4
#define HDC  64
#define EC   8
#define FFC  2048
#define VC   32000
#define LC   2
#define KVDIM (KVHC*HDC)   // 256

// ---------------------------------------------------------------- WMMA helpers
__device__ __forceinline__ f32x8 wmma_bf16(bf16x16 a, bf16x16 b, f32x8 c) {
  return __builtin_amdgcn_wmma_f32_16x16x32_bf16(false, a, false, b, (short)0, c,
                                                 false, false);
}

// fp32-source fragments (used by attention only)
// A-frag (16x32): element j holds K = j + 16*half
__device__ __forceinline__ bf16x16 load_a_frag_f32(const float* __restrict__ p, int half) {
  bf16x16 a;
#pragma unroll
  for (int j = 0; j < 8; ++j) a[j] = (__bf16)p[16 * half + j];
#pragma unroll
  for (int j = 0; j < 8; ++j) a[8 + j] = (__bf16)p[16 * half + 8 + j];
  return a;
}
// B-frag (32x16 of W^T, W row-major): element j holds K = j + 16*half
__device__ __forceinline__ bf16x16 load_b_frag_f32(const float* __restrict__ p, int half) {
  bf16x16 b;
#pragma unroll
  for (int j = 0; j < 16; ++j) b[j] = (__bf16)p[16 * half + j];
  return b;
}

// bf16-source fragments (pure vector loads, no cvt)
__device__ __forceinline__ bf16x16 load_a_frag_bf(const __bf16* __restrict__ p, int half) {
  bf16x8 lo = *(const bf16x8*)(p + 16 * half);         // 16B
  bf16x8 hi = *(const bf16x8*)(p + 16 * half + 8);    // 16B
  return __builtin_shufflevector(lo, hi, 0, 1, 2, 3, 4, 5, 6, 7,
                                 8, 9, 10, 11, 12, 13, 14, 15);
}
__device__ __forceinline__ bf16x16 load_b_frag_bf(const __bf16* __restrict__ p, int half) {
  return *(const bf16x16*)(p + 16 * half);            // 32B
}

// ---------------------------------------------------------------- fp32 -> bf16 pass
__global__ void cvt_bf16_kernel(const float* __restrict__ in, __bf16* __restrict__ out) {
  const size_t i = ((size_t)blockIdx.x * blockDim.x + threadIdx.x) * 4;
  const float4 v = *(const float4*)(in + i);
  bf16x4 o;
  o[0] = (__bf16)v.x; o[1] = (__bf16)v.y; o[2] = (__bf16)v.z; o[3] = (__bf16)v.w;
  *(bf16x4*)(out + i) = o;
}

// ---------------------------------------------------------------- NT GEMM, bf16 in / f32 out
// C[m][n] = sum_k A[m*lda+k] * W[n*ldw+k]; one wave computes a 64x32 tile
// (4 M-subtiles x 2 N-subtiles -> 8 f32x8 accumulators, no spills).
// grid = (N/32, M/64), block = 32.
__global__ __launch_bounds__(32)
void gemm_nt_bf16(const __bf16* __restrict__ A, const __bf16* __restrict__ W,
                  float* __restrict__ C, int K, int lda, int ldw, int ldc) {
  const int nb   = blockIdx.x * 32;
  const int mb   = blockIdx.y * 64;
  const int lane = threadIdx.x & 31;
  const int half = lane >> 4;
  const int r    = lane & 15;

  f32x8 acc[4][2];
#pragma unroll
  for (int i = 0; i < 4; ++i)
#pragma unroll
    for (int j = 0; j < 2; ++j) acc[i][j] = {};

  const __bf16* Ar[4];
  const __bf16* Wr[2];
#pragma unroll
  for (int i = 0; i < 4; ++i) Ar[i] = A + (size_t)(mb + 16 * i + r) * lda;
#pragma unroll
  for (int j = 0; j < 2; ++j) Wr[j] = W + (size_t)(nb + 16 * j + r) * ldw;

  for (int k0 = 0; k0 < K; k0 += 32) {
    bf16x16 af[4], wf[2];
#pragma unroll
    for (int i = 0; i < 4; ++i) af[i] = load_a_frag_bf(Ar[i] + k0, half);
#pragma unroll
    for (int j = 0; j < 2; ++j) wf[j] = load_b_frag_bf(Wr[j] + k0, half);
#pragma unroll
    for (int i = 0; i < 4; ++i)
#pragma unroll
      for (int j = 0; j < 2; ++j) acc[i][j] = wmma_bf16(af[i], wf[j], acc[i][j]);
  }

#pragma unroll
  for (int i = 0; i < 4; ++i)
#pragma unroll
    for (int j = 0; j < 2; ++j)
#pragma unroll
      for (int e = 0; e < 8; ++e) {
        const int m0 = mb + 16 * i + e + 8 * half;
        C[(size_t)m0 * ldc + nb + 16 * j + r] = acc[i][j][e];
      }
}

// ---------------------------------------------------------------- embed (+aux init)
__global__ void embed_kernel(const int* __restrict__ tokens, const float* __restrict__ emb,
                             float* __restrict__ x, float* __restrict__ aux) {
  const int t   = blockIdx.x;
  const int tok = tokens[t];
  const float* src = emb + (size_t)tok * DC;
  float* dst = x + (size_t)t * DC;
  for (int d = threadIdx.x; d < DC; d += blockDim.x) dst[d] = src[d];
  if (t == 0 && threadIdx.x == 0) *aux = 0.f;
}

// ---------------------------------------------------------------- rmsnorm (L2 form)
__global__ void rmsnorm_kernel(const float* __restrict__ x, const float* __restrict__ w,
                               float* __restrict__ y) {
  const int t = blockIdx.x;
  const float* xr = x + (size_t)t * DC;
  __shared__ float red[8];
  __shared__ float nrm;
  float s = 0.f;
  for (int d = threadIdx.x; d < DC; d += blockDim.x) { float v = xr[d]; s += v * v; }
  for (int off = 16; off; off >>= 1) s += __shfl_down(s, off, 32);
  if ((threadIdx.x & 31) == 0) red[threadIdx.x >> 5] = s;
  __syncthreads();
  if (threadIdx.x == 0) {
    float tot = 0.f;
    for (int i = 0; i < 8; ++i) tot += red[i];
    nrm = sqrtf(tot);
  }
  __syncthreads();
  const float inv = (1.0f / fmaxf(nrm, 1e-6f)) * 32.0f;  // sqrt(D)=32
  float* yr = y + (size_t)t * DC;
  for (int d = threadIdx.x; d < DC; d += blockDim.x) yr[d] = xr[d] * inv * w[d];
}

// ---------------------------------------------------------------- RoPE (q and k)
__global__ void rope_kernel(float* __restrict__ q, float* __restrict__ k) {
  const int t = blockIdx.x;
  const int s = t % SC;
  const int idx = threadIdx.x;  // 0 .. (HC+KVHC)*HDC/2 - 1 = 639
  float* base;
  int i;
  if (idx < HC * (HDC / 2)) {
    const int h = idx >> 5; i = idx & 31;
    base = q + (size_t)t * DC + h * HDC + 2 * i;
  } else {
    const int j = idx - HC * (HDC / 2);
    const int h = j >> 5; i = j & 31;
    base = k + (size_t)t * KVDIM + h * HDC + 2 * i;
  }
  const float freq = __expf(-logf(10000.0f) * (2.0f * (float)i) / (float)HDC);
  const float ang = (float)s * freq;
  float sn, cs;
  __sincosf(ang, &sn, &cs);
  const float x0 = base[0], x1 = base[1];
  base[0] = x0 * cs - x1 * sn;
  base[1] = x0 * sn + x1 * cs;
}

// ---------------------------------------------------------------- fused attention
// one wave per (b, h, 16-query block); scores tile in 64KB LDS
__global__ void attn_kernel(const float* __restrict__ q, const float* __restrict__ kk,
                            const float* __restrict__ v, float* __restrict__ o) {
  __shared__ float sc[16 * SC];  // 64KB
  const int nqb = SC / 16;
  const int qb  = blockIdx.x % nqb;
  const int h   = (blockIdx.x / nqb) % HC;
  const int b   = blockIdx.x / (nqb * HC);
  const int lane = threadIdx.x & 31;
  const int half = lane >> 4;
  const int r    = lane & 15;
  const int kvh  = h >> 2;  // H/KVH = 4

  // Q fragments (rows = queries, K-dim = head dim 64 -> two 32-chunks)
  const float* qrow = q + (size_t)(b * SC + qb * 16 + r) * DC + h * HDC;
  bf16x16 aq0 = load_a_frag_f32(qrow, half);
  bf16x16 aq1 = load_a_frag_f32(qrow + 32, half);

  const int nkb = qb + 1;  // valid key blocks of 16 (WINDOW >= S -> pure causal)

  // scores = (Q K^T)/8 with causal mask -> LDS
  for (int kb = 0; kb < nkb; ++kb) {
    const float* krow = kk + (size_t)(b * SC + kb * 16 + r) * KVDIM + kvh * HDC;
    bf16x16 bk0 = load_b_frag_f32(krow, half);
    bf16x16 bk1 = load_b_frag_f32(krow + 32, half);
    f32x8 cs = {};
    cs = wmma_bf16(aq0, bk0, cs);
    cs = wmma_bf16(aq1, bk1, cs);
    const int key = kb * 16 + r;
#pragma unroll
    for (int i = 0; i < 8; ++i) {
      const int m = i + 8 * half;            // local query row
      float val = cs[i] * 0.125f;            // 1/sqrt(64)
      if (key > qb * 16 + m) val = -1e30f;   // causal
      sc[m * SC + key] = val;
    }
  }
  __syncthreads();

  // softmax per row (lanes 0..15, one row each)
  const int len = nkb * 16;
  if (lane < 16) {
    float* row = &sc[lane * SC];
    float mx = -1e30f;
    for (int j = 0; j < len; ++j) mx = fmaxf(mx, row[j]);
    float sum = 0.f;
    for (int j = 0; j < len; ++j) { float e = __expf(row[j] - mx); row[j] = e; sum += e; }
    const float inv = 1.0f / sum;
    for (int j = 0; j < len; ++j) row[j] *= inv;
    if (nkb & 1)  // zero-pad to a multiple of 32 keys for the WMMA K-loop
      for (int j = len; j < len + 16; ++j) row[j] = 0.f;
  }
  __syncthreads();

  // O = P V  (A = P from LDS, B = V^T gathered from global)
  f32x8 oc[4] = {{}, {}, {}, {}};
  const int nk32 = (nkb + 1) >> 1;
  for (int kc = 0; kc < nk32; ++kc) {
    const int k0 = kc * 32;
    bf16x16 ap = load_a_frag_f32(&sc[r * SC + k0], half);
#pragma unroll
    for (int hc = 0; hc < 4; ++hc) {
      bf16x16 bv;
#pragma unroll
      for (int j = 0; j < 16; ++j) {
        const int key = k0 + 16 * half + j;
        bv[j] = (__bf16)v[(size_t)(b * SC + key) * KVDIM + kvh * HDC + hc * 16 + r];
      }
      oc[hc] = wmma_bf16(ap, bv, oc[hc]);
    }
  }
#pragma unroll
  for (int hc = 0; hc < 4; ++hc)
#pragma unroll
    for (int i = 0; i < 8; ++i) {
      const int m = i + 8 * half;
      o[(size_t)(b * SC + qb * 16 + m) * DC + h * HDC + hc * 16 + r] = oc[hc][i];
    }
}

// ---------------------------------------------------------------- router (1 wave/token)
__global__ void router_kernel(const float* __restrict__ xn, const float* __restrict__ rw,
                              float* __restrict__ probs, float* __restrict__ mask,
                              float* __restrict__ lse) {
  const int t = blockIdx.x;
  const int lane = threadIdx.x;
  float acc[EC];
#pragma unroll
  for (int e = 0; e < EC; ++e) acc[e] = 0.f;
  const float* xr = xn + (size_t)t * DC;
  for (int d = lane; d < DC; d += 32) {
    const float xv = xr[d];
#pragma unroll
    for (int e = 0; e < EC; ++e) acc[e] += xv * rw[e * DC + d];
  }
#pragma unroll
  for (int e = 0; e < EC; ++e)
    for (int off = 16; off; off >>= 1) acc[e] += __shfl_down(acc[e], off, 32);
  if (lane == 0) {
    float mx = acc[0];
    for (int e = 1; e < EC; ++e) mx = fmaxf(mx, acc[e]);
    float ex[EC], sum = 0.f;
    for (int e = 0; e < EC; ++e) { ex[e] = __expf(acc[e] - mx); sum += ex[e]; }
    lse[t] = mx + logf(sum);
    for (int e = 0; e < EC; ++e) probs[t * EC + e] = ex[e] / sum;
    int i1 = 0;
    for (int e = 1; e < EC; ++e) if (acc[e] > acc[i1]) i1 = e;
    int i2 = (i1 == 0) ? 1 : 0;
    for (int e = 0; e < EC; ++e) if (e != i1 && acc[e] > acc[i2]) i2 = e;
    for (int e = 0; e < EC; ++e) mask[t * EC + e] = (e == i1 || e == i2) ? 1.f : 0.f;
  }
}

// ---------------------------------------------------------------- aux losses
__global__ void aux_kernel(const float* __restrict__ lse, const float* __restrict__ probs,
                           float* __restrict__ aux) {
  __shared__ float red[3][8];
  float s_lse = 0.f, s1 = 0.f, s2 = 0.f;
  for (int t = threadIdx.x; t < TC; t += blockDim.x) s_lse += lse[t];
  const int N = SC * EC;
  for (int i = threadIdx.x; i < N; i += blockDim.x) {
    const int s = i >> 3, e = i & 7;
    const float ld = probs[s * EC + e] + probs[(SC + s) * EC + e];  // sum over batch
    s1 += ld; s2 += ld * ld;
  }
  for (int off = 16; off; off >>= 1) {
    s_lse += __shfl_down(s_lse, off, 32);
    s1 += __shfl_down(s1, off, 32);
    s2 += __shfl_down(s2, off, 32);
  }
  if ((threadIdx.x & 31) == 0) {
    const int w = threadIdx.x >> 5;
    red[0][w] = s_lse; red[1][w] = s1; red[2][w] = s2;
  }
  __syncthreads();
  if (threadIdx.x == 0) {
    float a = 0.f, bsum = 0.f, c = 0.f;
    for (int i = 0; i < 8; ++i) { a += red[0][i]; bsum += red[1][i]; c += red[2][i]; }
    float zl = a / (float)TC; zl = zl * zl;
    const float mean = bsum / (float)N;
    const float var = (c - (float)N * mean * mean) / (float)(N - 1);  // ddof=1
    const float ll = var / (mean * mean);                             // (std/mean)^2
    *aux += 0.001f * zl + 0.001f * ll;
  }
}

// ---------------------------------------------------------------- elementwise
__global__ void add_scaled_kernel(float* __restrict__ x, const float* __restrict__ y,
                                  float a) {
  const size_t i = (size_t)blockIdx.x * blockDim.x + threadIdx.x;
  x[i] += a * y[i];
}
__global__ void silu_mul_kernel(float* __restrict__ u, const float* __restrict__ u2) {
  const size_t i = (size_t)blockIdx.x * blockDim.x + threadIdx.x;
  const float a = u[i];
  u[i] = (a / (1.f + __expf(-a))) * u2[i];
}
__global__ void masked_add_kernel(float* __restrict__ x, const float* __restrict__ y,
                                  const float* __restrict__ mask, int e) {
  const size_t i = (size_t)blockIdx.x * blockDim.x + threadIdx.x;
  const int t = (int)(i / DC);
  x[i] += y[i] * mask[t * EC + e];
}

// ---------------------------------------------------------------- launcher
extern "C" void kernel_launch(void* const* d_in, const int* in_sizes, int n_in,
                              void* d_out, int out_size, void* d_ws, size_t ws_size,
                              hipStream_t stream) {
  (void)in_sizes; (void)n_in; (void)out_size; (void)ws_size;
  const int*   tokens   = (const int*)d_in[0];
  const float* tok_emb  = (const float*)d_in[1];
  const float* norm1_w  = (const float*)d_in[2];
  const float* norm2_w  = (const float*)d_in[3];
  const float* wq       = (const float*)d_in[4];
  const float* wk       = (const float*)d_in[5];
  const float* wv       = (const float*)d_in[6];
  const float* wo       = (const float*)d_in[7];
  const float* router_w = (const float*)d_in[8];
  const float* w1       = (const float*)d_in[9];
  const float* w2       = (const float*)d_in[10];
  const float* w3       = (const float*)d_in[11];
  const float* normf_w  = (const float*)d_in[12];
  const float* head_w   = (const float*)d_in[13];

  float* out = (float*)d_out;                       // (T, V) logits
  float* aux = out + (size_t)TC * VC;               // trailing scalar

  const size_t TD  = (size_t)TC * DC;
  const size_t TKV = (size_t)TC * KVDIM;
  const size_t TFF = (size_t)TC * FFC;
  float* ws   = (float*)d_ws;
  float* x    = ws;                 ws += TD;
  float* xn   = ws;                 ws += TD;
  float* qb   = ws;                 ws += TD;
  float* kb   = ws;                 ws += TKV;
  float* vb   = ws;                 ws += TKV;
  float* ob   = ws;                 ws += TD;
  float* yb   = ws;                 ws += TD;
  float* t1   = ws;                 ws += TFF;
  float* t2   = ws;                 ws += TFF;
  float* rprob = ws;                ws += (size_t)TC * EC;
  float* rmask = ws;                ws += (size_t)TC * EC;
  float* rlse  = ws;                ws += TC;
  __bf16* xnb  = (__bf16*)ws;       ws += TD / 2;            // (T,D) bf16
  __bf16* abuf = (__bf16*)ws;       ws += TFF / 2;           // (T,FF) bf16
  __bf16* wbuf = (__bf16*)ws;       ws += ((size_t)VC * DC) / 2;  // largest weight, bf16

  auto cvt = [&](const float* in, __bf16* o, size_t n) {  // n % 1024 == 0
    hipLaunchKernelGGL(cvt_bf16_kernel, dim3((unsigned)(n / 1024)), dim3(256), 0, stream,
                       in, o);
  };
  auto gemm = [&](const __bf16* A, const __bf16* W, float* C, int M, int N, int K,
                  int lda, int ldw, int ldc) {
    dim3 grid(N / 32, M / 64);
    hipLaunchKernelGGL(gemm_nt_bf16, grid, dim3(32), 0, stream, A, W, C, K, lda, ldw, ldc);
  };

  hipLaunchKernelGGL(embed_kernel, dim3(TC), dim3(256), 0, stream, tokens, tok_emb, x, aux);

  for (int l = 0; l < LC; ++l) {
    // ---- attention block
    hipLaunchKernelGGL(rmsnorm_kernel, dim3(TC), dim3(256), 0, stream,
                       x, norm1_w + (size_t)l * DC, xn);
    cvt(xn, xnb, TD);
    cvt(wq + (size_t)l * DC * DC, wbuf, (size_t)DC * DC);
    gemm(xnb, wbuf, qb, TC, DC, DC, DC, DC, DC);
    cvt(wk + (size_t)l * KVDIM * DC, wbuf, (size_t)KVDIM * DC);
    gemm(xnb, wbuf, kb, TC, KVDIM, DC, DC, DC, KVDIM);
    cvt(wv + (size_t)l * KVDIM * DC, wbuf, (size_t)KVDIM * DC);
    gemm(xnb, wbuf, vb, TC, KVDIM, DC, DC, DC, KVDIM);
    hipLaunchKernelGGL(rope_kernel, dim3(TC), dim3((HC + KVHC) * HDC / 2), 0, stream, qb, kb);
    hipLaunchKernelGGL(attn_kernel, dim3(BC * HC * (SC / 16)), dim3(32), 0, stream,
                       qb, kb, vb, ob);
    cvt(ob, abuf, TD);
    cvt(wo + (size_t)l * DC * DC, wbuf, (size_t)DC * DC);
    gemm(abuf, wbuf, yb, TC, DC, DC, DC, DC, DC);
    hipLaunchKernelGGL(add_scaled_kernel, dim3((unsigned)(TD / 256)), dim3(256), 0, stream,
                       x, yb, 0.5f);  // 1/sqrt(2L)

    // ---- MoE block
    hipLaunchKernelGGL(rmsnorm_kernel, dim3(TC), dim3(256), 0, stream,
                       x, norm2_w + (size_t)l * DC, xn);
    cvt(xn, xnb, TD);
    hipLaunchKernelGGL(router_kernel, dim3(TC), dim3(32), 0, stream,
                       xn, router_w + (size_t)l * EC * DC, rprob, rmask, rlse);
    hipLaunchKernelGGL(aux_kernel, dim3(1), dim3(256), 0, stream, rlse, rprob, aux);

    for (int e = 0; e < EC; ++e) {
      const float* w1e = w1 + ((size_t)(l * EC + e)) * FFC * DC;
      const float* w2e = w2 + ((size_t)(l * EC + e)) * FFC * DC;
      const float* w3e = w3 + ((size_t)(l * EC + e)) * DC * FFC;
      cvt(w1e, wbuf, (size_t)FFC * DC);
      gemm(xnb, wbuf, t1, TC, FFC, DC, DC, DC, FFC);
      cvt(w2e, wbuf, (size_t)FFC * DC);
      gemm(xnb, wbuf, t2, TC, FFC, DC, DC, DC, FFC);
      hipLaunchKernelGGL(silu_mul_kernel, dim3((unsigned)(TFF / 256)), dim3(256), 0, stream,
                         t1, t2);
      cvt(t1, abuf, TFF);
      cvt(w3e, wbuf, (size_t)DC * FFC);
      gemm(abuf, wbuf, yb, TC, DC, FFC, FFC, FFC, DC);
      hipLaunchKernelGGL(masked_add_kernel, dim3((unsigned)(TD / 256)), dim3(256), 0, stream,
                         x, yb, rmask, e);
    }
  }

  hipLaunchKernelGGL(rmsnorm_kernel, dim3(TC), dim3(256), 0, stream, x, normf_w, xn);
  cvt(xn, xnb, TD);
  cvt(head_w, wbuf, (size_t)VC * DC);
  gemm(xnb, wbuf, out, TC, VC, DC, DC, DC, VC);
}
